// TokenPacker_70806830842116
// MI455X (gfx1250) — compile-verified, matrix operand order checked by reading, not run on hardware
//
#include <hip/hip_runtime.h>
#include <math.h>

// ---------------- problem constants ----------------
#define C_DIM 1024
#define N_HEADS 8
#define HEAD_D 128
#define TOK_ALL 81920   // N*D*H*W = 2*32*40*32
#define TOK_Q   10240   // N*dg*hg*wg = 2*16*20*16
#define XD 32
#define XH 40
#define XW 32
#define QDG 16
#define QHG 20
#define QWG 16

typedef __attribute__((ext_vector_type(16))) __bf16 v16bf;
typedef __attribute__((ext_vector_type(8)))  float  v8f;

// ---------------- scalar bf16 helpers ----------------
__device__ __forceinline__ unsigned short f2bf_u(float f) {
  union { float f; unsigned u; } v; v.f = f;
  unsigned u = v.u;
  unsigned r = u + 0x7FFFu + ((u >> 16) & 1u);   // round-to-nearest-even
  return (unsigned short)(r >> 16);
}
__device__ __forceinline__ float bf2f(unsigned short h) {
  union { unsigned u; float f; } v; v.u = ((unsigned)h) << 16;
  return v.f;
}
__device__ __forceinline__ float gelu_exact(float x) {
  return 0.5f * x * (1.0f + erff(x * 0.70710678118654752f));
}

union FragU {
  v16bf v;
  unsigned short u[16];
  uint4 q[2];
};

// Load a WMMA 16-bit fragment: 8 contiguous bf16 at p, plus 8 at p+16.
// (Matches ISA 16-bit A/B lane layout: lanes 0-15 hold K {0..7,16..23},
//  lanes 16-31 hold K {8..15,24..31}; caller offsets p by half*8.)
__device__ __forceinline__ v16bf load_frag(const unsigned short* p) {
  FragU t;
  t.q[0] = *(const uint4*)(p);
  t.q[1] = *(const uint4*)(p + 16);
  return t.v;
}
__device__ __forceinline__ v16bf load_frag(const float* p) {
  FragU t;
#pragma unroll
  for (int i = 0; i < 8; ++i) t.u[i] = f2bf_u(p[i]);
#pragma unroll
  for (int i = 0; i < 8; ++i) t.u[8 + i] = f2bf_u(p[16 + i]);
  return t.v;
}

__device__ __forceinline__ void store_out(float* p, float v) { *p = v; }
__device__ __forceinline__ void store_out(unsigned short* p, float v) { *p = f2bf_u(v); }

__device__ __forceinline__ void load4bf(float* o, const unsigned short* p) {
  uint2 r = *(const uint2*)(p);
  o[0] = bf2f((unsigned short)(r.x & 0xffffu));
  o[1] = bf2f((unsigned short)(r.x >> 16));
  o[2] = bf2f((unsigned short)(r.y & 0xffffu));
  o[3] = bf2f((unsigned short)(r.y >> 16));
}
__device__ __forceinline__ void store4bf(unsigned short* p, const float* a) {
  uint2 r;
  r.x = (unsigned)f2bf_u(a[0]) | ((unsigned)f2bf_u(a[1]) << 16);
  r.y = (unsigned)f2bf_u(a[2]) | ((unsigned)f2bf_u(a[3]) << 16);
  *(uint2*)(p) = r;
}

// ---------------- fp32 -> bf16 weight convert ----------------
__global__ __launch_bounds__(256) void cvt_f32_bf16(const float* __restrict__ in,
                                                    unsigned short* __restrict__ out,
                                                    int n) {
  int i = blockIdx.x * 256 + threadIdx.x;
  if (i < n) out[i] = f2bf_u(in[i]);
}

// ---------------- WMMA GEMM: out = act(A @ W^T + bias) ----------------
// A:[M,K] (float or bf16, row-major), W:[N,K] bf16 row-major, out:[M,N]
// EPI: 0 = linear, 1 = exact GELU
// Block = 256 thr = 8 waves; wave computes a 32(M) x 64(N) tile
// (2 A fragments x 4 B fragments -> 8 WMMAs per K-step, B reused 2x).
template <int EPI, typename TIN, typename TOUT>
__global__ __launch_bounds__(256) void gemm_bias_act(
    const TIN* __restrict__ A, const unsigned short* __restrict__ W,
    const float* __restrict__ bias, TOUT* __restrict__ out,
    int M, int N, int K) {
  const int lane = threadIdx.x & 31;
  const int wave = threadIdx.x >> 5;
  const int mTile = blockIdx.x * 256 + wave * 32;
  const int nTile = blockIdx.y * 64;
  if (mTile >= M) return;
  const int half = lane >> 4;        // K-half selector per ISA A-layout
  const int l15  = lane & 15;        // row (A) / col (B) owned by this lane
  const size_t rowA0 = (size_t)(mTile + l15) * (size_t)K;
  const size_t rowA1 = (size_t)(mTile + 16 + l15) * (size_t)K;
  const size_t rowB  = (size_t)(nTile + l15) * (size_t)K;

  v8f acc[2][4] = {};
  for (int kk = 0; kk < K; kk += 32) {
    const int kbase = kk + half * 8; // lanes 16-31 take K 8..15 / 24..31
    v16bf a0 = load_frag(A + rowA0 + kbase);
    v16bf a1 = load_frag(A + rowA1 + kbase);
    if (kk + 32 < K) {
      // cover next K-step's lines (lowers to global_prefetch_b8)
      __builtin_prefetch(A + rowA0 + kbase + 32, 0, 3);
      __builtin_prefetch(A + rowA1 + kbase + 32, 0, 3);
      __builtin_prefetch(W + rowB + kbase + 32, 0, 3);
    }
#pragma unroll
    for (int t = 0; t < 4; ++t) {
      v16bf b = load_frag(W + rowB + (size_t)t * 16 * K + kbase);
      acc[0][t] = __builtin_amdgcn_wmma_f32_16x16x32_bf16(
          false, a0, false, b, (short)0, acc[0][t], false, false);
      acc[1][t] = __builtin_amdgcn_wmma_f32_16x16x32_bf16(
          false, a1, false, b, (short)0, acc[1][t], false, false);
    }
  }
  // C/D layout: VGPR r -> M = mTile + u*16 + r + 8*half ; N = nTile + t*16 + l15
#pragma unroll
  for (int u = 0; u < 2; ++u) {
#pragma unroll
    for (int t = 0; t < 4; ++t) {
      const int n = nTile + t * 16 + l15;
      const float b = bias ? bias[n] : 0.0f;
#pragma unroll
      for (int r = 0; r < 8; ++r) {
        const int m = mTile + u * 16 + r + 8 * half;
        float v = acc[u][t][r] + b;
        if (EPI == 1) v = gelu_exact(v);
        store_out(out + (size_t)m * (size_t)N + n, v);
      }
    }
  }
}

// ---------------- LayerNorm over C=1024, in-place on bf16 ----------------
__global__ __launch_bounds__(256) void layernorm_bf16(
    unsigned short* __restrict__ buf, const float* __restrict__ gamma,
    const float* __restrict__ beta) {
  unsigned short* p = buf + (size_t)blockIdx.x * C_DIM;
  const int i0 = threadIdx.x * 4;
  float x[4];
  load4bf(x, p + i0);
  float s  = x[0] + x[1] + x[2] + x[3];
  float ss = x[0]*x[0] + x[1]*x[1] + x[2]*x[2] + x[3]*x[3];
  __shared__ float rs[256], rq[256];
  rs[threadIdx.x] = s; rq[threadIdx.x] = ss;
  __syncthreads();
  for (int off = 128; off > 0; off >>= 1) {
    if (threadIdx.x < off) {
      rs[threadIdx.x] += rs[threadIdx.x + off];
      rq[threadIdx.x] += rq[threadIdx.x + off];
    }
    __syncthreads();
  }
  const float mean = rs[0] * (1.0f / C_DIM);
  const float var  = rq[0] * (1.0f / C_DIM) - mean * mean;
  const float inv  = rsqrtf(var + 1e-6f);
  float y[4];
#pragma unroll
  for (int i = 0; i < 4; ++i)
    y[i] = (x[i] - mean) * inv * gamma[i0 + i] + beta[i0 + i];
  store4bf(p + i0, y);
}

// ---------------- 2x2x2 average pool (x fp32 -> bf16 q_pool) ----------------
__global__ __launch_bounds__(256) void pool_kernel(const float* __restrict__ x,
                                                   unsigned short* __restrict__ qp) {
  size_t idx = (size_t)blockIdx.x * 256 + threadIdx.x;
  if (idx >= (size_t)TOK_Q * C_DIM) return;
  int c = (int)(idx & (C_DIM - 1));
  int q = (int)(idx >> 10);
  int nb = q / (QDG * QHG * QWG);
  int r  = q % (QDG * QHG * QWG);
  int dgi = r / (QHG * QWG);
  int rem = r % (QHG * QWG);
  int hgi = rem / QWG;
  int wgi = rem % QWG;
  float s = 0.0f;
#pragma unroll
  for (int j = 0; j < 8; ++j) {
    int dd = dgi * 2 + ((j >> 2) & 1);
    int hh = hgi * 2 + ((j >> 1) & 1);
    int ww = wgi * 2 + (j & 1);
    size_t tok = (size_t)nb * (XD * XH * XW) + ((size_t)dd * XH + hh) * XW + ww;
    s += x[tok * C_DIM + c];
  }
  qp[idx] = f2bf_u(s * 0.125f);
}

// ---------------- attention: 8 keys per query, 8 heads ----------------
// grid = TOK_Q blocks, block = 256 = 8 waves; wave w handles head w.
__global__ __launch_bounds__(256) void attn_kernel(
    const unsigned short* __restrict__ qh, const unsigned short* __restrict__ kh,
    const unsigned short* __restrict__ vh, unsigned short* __restrict__ o) {
  const int q = blockIdx.x;
  const int h = threadIdx.x >> 5;
  const int lane = threadIdx.x & 31;

  int nb = q / (QDG * QHG * QWG);
  int r  = q % (QDG * QHG * QWG);
  int dgi = r / (QHG * QWG);
  int rem = r % (QHG * QWG);
  int hgi = rem / QWG;
  int wgi = rem % QWG;
  int toks[8];
#pragma unroll
  for (int j = 0; j < 8; ++j) {
    int dd = dgi * 2 + ((j >> 2) & 1);
    int hh = hgi * 2 + ((j >> 1) & 1);
    int ww = wgi * 2 + (j & 1);
    toks[j] = nb * (XD * XH * XW) + (dd * XH + hh) * XW + ww;
  }
  const int base = h * HEAD_D + lane * 4;
  float qv[4];
  load4bf(qv, &qh[(size_t)q * C_DIM + base]);

  float scores[8];
#pragma unroll
  for (int j = 0; j < 8; ++j) {
    float kv[4];
    load4bf(kv, &kh[(size_t)toks[j] * C_DIM + base]);
    float d = qv[0]*kv[0] + qv[1]*kv[1] + qv[2]*kv[2] + qv[3]*kv[3];
#pragma unroll
    for (int m = 16; m >= 1; m >>= 1) d += __shfl_xor(d, m, 32);
    scores[j] = d * 0.0883883476483184f;   // 1/sqrt(128)
  }
  float mx = scores[0];
#pragma unroll
  for (int j = 1; j < 8; ++j) mx = fmaxf(mx, scores[j]);
  float e[8], sum = 0.0f;
#pragma unroll
  for (int j = 0; j < 8; ++j) { e[j] = __expf(scores[j] - mx); sum += e[j]; }
  const float rinv = 1.0f / sum;

  float acc[4] = {0.f, 0.f, 0.f, 0.f};
#pragma unroll
  for (int j = 0; j < 8; ++j) {
    float vv[4];
    load4bf(vv, &vh[(size_t)toks[j] * C_DIM + base]);
    float w = e[j] * rinv;
#pragma unroll
    for (int i = 0; i < 4; ++i) acc[i] += w * vv[i];
  }
  store4bf(&o[(size_t)q * C_DIM + base], acc);
}

// ---------------- host launch ----------------
extern "C" void kernel_launch(void* const* d_in, const int* in_sizes, int n_in,
                              void* d_out, int out_size, void* d_ws, size_t ws_size,
                              hipStream_t stream) {
  (void)in_sizes; (void)n_in; (void)out_size; (void)ws_size;
  const float* x      = (const float*)d_in[0];
  const float* q_w    = (const float*)d_in[1];
  const float* k_w1   = (const float*)d_in[2];
  const float* k_b1   = (const float*)d_in[3];
  const float* k_w2   = (const float*)d_in[4];
  const float* k_b2   = (const float*)d_in[5];
  const float* v_w1   = (const float*)d_in[6];
  const float* v_b1   = (const float*)d_in[7];
  const float* v_w2   = (const float*)d_in[8];
  const float* v_b2   = (const float*)d_in[9];
  const float* lnq_w  = (const float*)d_in[10];
  const float* lnq_b  = (const float*)d_in[11];
  const float* lnk_w  = (const float*)d_in[12];
  const float* lnk_b  = (const float*)d_in[13];
  const float* lnv_w  = (const float*)d_in[14];
  const float* lnv_b  = (const float*)d_in[15];
  const float* in_w   = (const float*)d_in[16];
  const float* in_b   = (const float*)d_in[17];
  const float* out_w  = (const float*)d_in[18];
  const float* out_b  = (const float*)d_in[19];
  const float* mlp_w1 = (const float*)d_in[20];
  const float* mlp_b1 = (const float*)d_in[21];
  const float* mlp_w2 = (const float*)d_in[22];
  const float* mlp_b2 = (const float*)d_in[23];

  char* ws = (char*)d_ws;
  size_t off = 0;
  auto alloc_us = [&](size_t nelem) -> unsigned short* {
    unsigned short* p = (unsigned short*)(ws + off);
    off += ((nelem * 2) + 255) & ~(size_t)255;
    return p;
  };
  const size_t CC = (size_t)C_DIM * C_DIM;   // 1,048,576

  // bf16 weight copies
  unsigned short* w_qw   = alloc_us(CC);
  unsigned short* w_kw1  = alloc_us(CC);
  unsigned short* w_kw2  = alloc_us(CC);
  unsigned short* w_vw1  = alloc_us(CC);
  unsigned short* w_vw2  = alloc_us(CC);
  unsigned short* w_inw  = alloc_us(3 * CC);
  unsigned short* w_outw = alloc_us(CC);
  unsigned short* w_m1   = alloc_us(2 * CC);
  unsigned short* w_m2   = alloc_us(4 * CC);

  // large activation buffers (bf16)
  const size_t BIG = (size_t)TOK_ALL * C_DIM;     // 83,886,080
  unsigned short* bufA = alloc_us(BIG);           // hidden, later kh
  unsigned short* bufB = alloc_us(BIG);           // k, later vh
  unsigned short* bufC = alloc_us(BIG);           // v
  // query-side buffers
  const size_t QS = (size_t)TOK_Q * C_DIM;        // 10,485,760
  unsigned short* qp   = alloc_us(QS);
  unsigned short* qln  = alloc_us(QS);
  unsigned short* qhb  = alloc_us(QS);
  unsigned short* ob   = alloc_us(QS);
  unsigned short* o2b  = alloc_us(QS);
  unsigned short* m1b  = alloc_us((size_t)TOK_Q * 2 * C_DIM);

  // --- convert weights to bf16 ---
  struct CvtJob { const float* src; unsigned short* dst; int n; };
  CvtJob jobs[9] = {
      {q_w, w_qw, (int)CC},       {k_w1, w_kw1, (int)CC},  {k_w2, w_kw2, (int)CC},
      {v_w1, w_vw1, (int)CC},     {v_w2, w_vw2, (int)CC},  {in_w, w_inw, (int)(3*CC)},
      {out_w, w_outw, (int)CC},   {mlp_w1, w_m1, (int)(2*CC)}, {mlp_w2, w_m2, (int)(4*CC)}};
  for (int i = 0; i < 9; ++i)
    cvt_f32_bf16<<<(jobs[i].n + 255) / 256, 256, 0, stream>>>(jobs[i].src, jobs[i].dst, jobs[i].n);

  const dim3 blk(256);
  const dim3 gBigC(TOK_ALL / 256, C_DIM / 64);      // 320 x 16
  const dim3 gQC(TOK_Q / 256, C_DIM / 64);          // 40 x 16
  const dim3 gQ2C(TOK_Q / 256, 2 * C_DIM / 64);     // 40 x 32

  // --- k branch: k = LN(gelu(x@k_w1^T+b1)@k_w2^T+b2) ---
  gemm_bias_act<1, float, unsigned short><<<gBigC, blk, 0, stream>>>(
      x, w_kw1, k_b1, bufA, TOK_ALL, C_DIM, C_DIM);
  gemm_bias_act<0, unsigned short, unsigned short><<<gBigC, blk, 0, stream>>>(
      bufA, w_kw2, k_b2, bufB, TOK_ALL, C_DIM, C_DIM);
  layernorm_bf16<<<TOK_ALL, blk, 0, stream>>>(bufB, lnk_w, lnk_b);

  // --- v branch ---
  gemm_bias_act<1, float, unsigned short><<<gBigC, blk, 0, stream>>>(
      x, w_vw1, v_b1, bufA, TOK_ALL, C_DIM, C_DIM);
  gemm_bias_act<0, unsigned short, unsigned short><<<gBigC, blk, 0, stream>>>(
      bufA, w_vw2, v_b2, bufC, TOK_ALL, C_DIM, C_DIM);
  layernorm_bf16<<<TOK_ALL, blk, 0, stream>>>(bufC, lnv_w, lnv_b);

  // --- head projections: kh = k@wk^T+bk -> bufA ; vh = v@wv^T+bv -> bufB ---
  gemm_bias_act<0, unsigned short, unsigned short><<<gBigC, blk, 0, stream>>>(
      bufB, w_inw + CC, in_b + C_DIM, bufA, TOK_ALL, C_DIM, C_DIM);
  gemm_bias_act<0, unsigned short, unsigned short><<<gBigC, blk, 0, stream>>>(
      bufC, w_inw + 2 * CC, in_b + 2 * C_DIM, bufB, TOK_ALL, C_DIM, C_DIM);

  // --- query: pool -> @q_w^T -> LN -> @wq^T+bq ---
  pool_kernel<<<(unsigned)((QS + 255) / 256), blk, 0, stream>>>(x, qp);
  gemm_bias_act<0, unsigned short, unsigned short><<<gQC, blk, 0, stream>>>(
      qp, w_qw, (const float*)nullptr, qln, TOK_Q, C_DIM, C_DIM);
  layernorm_bf16<<<TOK_Q, blk, 0, stream>>>(qln, lnq_w, lnq_b);
  gemm_bias_act<0, unsigned short, unsigned short><<<gQC, blk, 0, stream>>>(
      qln, w_inw, in_b, qhb, TOK_Q, C_DIM, C_DIM);

  // --- attention over 8-element regions ---
  attn_kernel<<<TOK_Q, blk, 0, stream>>>(qhb, bufA, bufB, ob);

  // --- output projection + MLP ---
  gemm_bias_act<0, unsigned short, unsigned short><<<gQC, blk, 0, stream>>>(
      ob, w_outw, out_b, o2b, TOK_Q, C_DIM, C_DIM);
  gemm_bias_act<1, unsigned short, unsigned short><<<gQ2C, blk, 0, stream>>>(
      o2b, w_m1, mlp_b1, m1b, TOK_Q, 2 * C_DIM, C_DIM);
  gemm_bias_act<0, unsigned short, float><<<gQ2C, blk, 0, stream>>>(
      m1b, w_m2, mlp_b2, (float*)d_out, TOK_Q, 2 * C_DIM, 2 * C_DIM);
}
